// GSAT_HeteroGNN_52707838656535
// MI455X (gfx1250) — matrix operand order repaired and checked by reading, not compiled, not run on hardware
//
#include <hip/hip_runtime.h>

// ---------------------------------------------------------------------------
// GSAT Hetero-GNN forward for gfx1250 (MI455X).
// GEMMs use V_WMMA_F32_16X16X32_BF16 (bf16 inputs, fp32 accumulate), with
// A-fragments register-resident across all 16 n-tiles of a 16x256 strip and an
// explicit double-buffered B-fragment stream (two clauses of global_load_b128
// in flight behind each v_wmma).
// Scatter-mean aggregation uses global_atomic_add_f32.
// ---------------------------------------------------------------------------

typedef __attribute__((ext_vector_type(16))) __bf16 v16bf;
typedef __attribute__((ext_vector_type(8)))  float  v8f;

#define NA_N 50000L
#define NP_N 100000L
#define IN_D 128
#define H_D  256

// ---------------------------- utility kernels ------------------------------

__global__ void k_zero_f32(float* __restrict__ p, long n) {
  long i = (long)blockIdx.x * blockDim.x + threadIdx.x;
  if (i < n) p[i] = 0.0f;
}

__global__ void k_f32_to_bf16(const float* __restrict__ in, __bf16* __restrict__ out, long n) {
  long i = (long)blockIdx.x * blockDim.x + threadIdx.x;
  if (i < n) out[i] = (__bf16)in[i];
}

__global__ void k_count(const int* __restrict__ dst, long E, float* __restrict__ cnt) {
  long e = (long)blockIdx.x * blockDim.x + threadIdx.x;
  if (e < E) atomicAdd(cnt + dst[e], 1.0f);
}

// thread handles 4 consecutive feature channels of one edge
__global__ void k_scatter_add(const void* __restrict__ X, int x_is_f32, int d,
                              const int* __restrict__ src, const int* __restrict__ dst,
                              long E, float* __restrict__ agg) {
  long t = (long)blockIdx.x * blockDim.x + threadIdx.x;
  int per = d >> 2;
  long e = t / per;
  if (e >= E) return;
  int c = (int)(t % per) << 2;
  int s = src[e], dd = dst[e];
  float v0, v1, v2, v3;
  if (x_is_f32) {
    const float* p = (const float*)X + (size_t)s * d + c;
    v0 = p[0]; v1 = p[1]; v2 = p[2]; v3 = p[3];
  } else {
    const __bf16* p = (const __bf16*)X + (size_t)s * d + c;
    v0 = (float)p[0]; v1 = (float)p[1]; v2 = (float)p[2]; v3 = (float)p[3];
  }
  float* o = agg + (size_t)dd * d + c;
  atomicAdd(o + 0, v0);
  atomicAdd(o + 1, v1);
  atomicAdd(o + 2, v2);
  atomicAdd(o + 3, v3);
}

__global__ void k_div_rows(float* __restrict__ agg, const float* __restrict__ cnt,
                           long M, int d) {
  long i = (long)blockIdx.x * blockDim.x + threadIdx.x;
  if (i >= M * (long)d) return;
  long r = i / d;
  agg[i] = agg[i] / fmaxf(cnt[r], 1.0f);
}

// ------------------------------ WMMA GEMM ----------------------------------
// out[M,256] = relu?( bias0 + bias1 + sum_s A_s[M,K_s] @ W_s[256,K_s]^T )
// One wave per 16-row strip; 16 n-tiles of 16 columns each.  All A fragments
// (up to 20 = 640/32) are loaded/converted once and kept in VGPRs; the inner
// loop streams B fragments through an explicit double buffer so the allocator
// must keep two fragments (two load clauses) in flight behind each v_wmma.
//
// bf16 A-fragment layout (16x32, ISA 7.12.2): lane<16 -> M=lane, K in
// {0..7,16..23}; lane>=16 -> M=lane-16, K in {8..15,24..31}.  Each half is 8
// contiguous bf16 = one 16B load.
// bf16 B-fragment layout (32x16): lane -> N=lane&15; K in {0..7,16..23} for
// lane<16, {8..15,24..31} for lane>=16 (same as A) = two 16B loads.

template <int K, int AF32>
__device__ __forceinline__ void load_a_frags(v16bf* f, const void* A, int m0, int lane) {
  if (K == 0) return;
  const int row  = lane & 15;
  const int half = lane >> 4;
#pragma unroll
  for (int kk = 0; kk < K / 32; ++kk) {
    const int k = kk * 32;
    v16bf a;
    if (AF32) {
      const float* pa = (const float*)A + (size_t)(m0 + row) * K + k + (half << 3);
      float4 l0 = *(const float4*)(pa);
      float4 l1 = *(const float4*)(pa + 4);
      float4 h0 = *(const float4*)(pa + 16);
      float4 h1 = *(const float4*)(pa + 20);
      a[0] = (__bf16)l0.x; a[1] = (__bf16)l0.y; a[2] = (__bf16)l0.z; a[3] = (__bf16)l0.w;
      a[4] = (__bf16)l1.x; a[5] = (__bf16)l1.y; a[6] = (__bf16)l1.z; a[7] = (__bf16)l1.w;
      a[8]  = (__bf16)h0.x; a[9]  = (__bf16)h0.y; a[10] = (__bf16)h0.z; a[11] = (__bf16)h0.w;
      a[12] = (__bf16)h1.x; a[13] = (__bf16)h1.y; a[14] = (__bf16)h1.z; a[15] = (__bf16)h1.w;
    } else {
      const __bf16* pa = (const __bf16*)A + (size_t)(m0 + row) * K + k + (half << 3);
      ((uint4*)&a)[0] = *(const uint4*)(pa);
      ((uint4*)&a)[1] = *(const uint4*)(pa + 16);
    }
    f[kk] = a;
  }
}

template <int K>
__device__ __forceinline__ v16bf load_b_frag(const __bf16* __restrict__ W,
                                             int nt, int kk, int lane) {
  const int row  = lane & 15;
  const int half = lane >> 4;
  const __bf16* pb = W + (size_t)(nt * 16 + row) * K + kk * 32 + (half << 3);
  v16bf b;
  ((uint4*)&b)[0] = *(const uint4*)(pb);
  ((uint4*)&b)[1] = *(const uint4*)(pb + 16);
  return b;
}

// explicit double-buffered K-loop for one (A,W) slot: b0/b1 are both live
// across every wmma, forcing two outstanding load clauses.
template <int K>
__device__ __forceinline__ void gemm_slot(v8f& acc, const v16bf* __restrict__ af,
                                          const __bf16* __restrict__ W,
                                          int nt, int lane) {
  constexpr int NF = K / 32;
  v16bf b0 = load_b_frag<K>(W, nt, 0, lane);
  v16bf b1;
  if (NF > 1) b1 = load_b_frag<K>(W, nt, 1, lane);
#pragma unroll
  for (int kk = 0; kk < NF; ++kk) {
    if ((kk & 1) == 0) {
      acc = __builtin_amdgcn_wmma_f32_16x16x32_bf16(false, af[kk], false, b0,
                                                    (short)0, acc, false, false);
      if (kk + 2 < NF) b0 = load_b_frag<K>(W, nt, kk + 2, lane);
    } else {
      acc = __builtin_amdgcn_wmma_f32_16x16x32_bf16(false, af[kk], false, b1,
                                                    (short)0, acc, false, false);
      if (kk + 2 < NF) b1 = load_b_frag<K>(W, nt, kk + 2, lane);
    }
  }
}

template <int K0, int F0, int K1, int F1, int K2, int F2>
__global__ __launch_bounds__(32)
void k_sage_gemm(const void* A0, const __bf16* __restrict__ W0,
                 const void* A1, const __bf16* __restrict__ W1,
                 const void* A2, const __bf16* __restrict__ W2,
                 const float* __restrict__ bias0, const float* __restrict__ bias1,
                 __bf16* __restrict__ out, int do_relu) {
  const int lane = threadIdx.x;
  const int m0   = blockIdx.x * 16;
  const int row  = lane & 15;
  const int half = lane >> 4;

  constexpr int NF0 = K0 / 32, NF1 = K1 / 32, NF2 = K2 / 32;
  v16bf af[NF0 + NF1 + NF2];
  load_a_frags<K0, F0>(af, A0, m0, lane);
  load_a_frags<K1, F1>(af + NF0, A1, m0, lane);
  load_a_frags<K2, F2>(af + NF0 + NF1, A2, m0, lane);

#pragma unroll 1
  for (int nt = 0; nt < 16; ++nt) {
    const int n = nt * 16 + row;
    float binit = bias0[n] + (bias1 ? bias1[n] : 0.0f);
    v8f acc;
#pragma unroll
    for (int i = 0; i < 8; ++i) acc[i] = binit;

    gemm_slot<K0>(acc, af, W0, nt, lane);
    if (K1 > 0) gemm_slot<K1>(acc, af + NF0, W1, nt, lane);
    if (K2 > 0) gemm_slot<K2>(acc, af + NF0 + NF1, W2, nt, lane);

    // D layout: VGPR v -> M = v + (lane>=16 ? 8 : 0), N = lane&15
    const int mb = m0 + (half << 3);
#pragma unroll
    for (int v = 0; v < 8; ++v) {
      float x = acc[v];
      if (do_relu) x = fmaxf(x, 0.0f);
      out[(size_t)(mb + v) * H_D + n] = (__bf16)x;
    }
  }
}

// ------------------------------ pooling / MLP ------------------------------

__global__ void k_col_mean(const __bf16* __restrict__ X, long M, float inv,
                           float* __restrict__ out) {
  __shared__ float sm[256];
  int j = blockIdx.x; // column 0..255
  float s = 0.0f;
  for (long i = threadIdx.x; i < M; i += blockDim.x)
    s += (float)X[(size_t)i * H_D + j];
  sm[threadIdx.x] = s;
  __syncthreads();
  for (int w = 128; w > 0; w >>= 1) {
    if ((int)threadIdx.x < w) sm[threadIdx.x] += sm[threadIdx.x + w];
    __syncthreads();
  }
  if (threadIdx.x == 0) out[j] = sm[0] * inv;
}

__global__ void k_mlp(const float* __restrict__ pooled, // [512]
                      const float* __restrict__ W1, const float* __restrict__ b1,
                      const float* __restrict__ W2, const float* __restrict__ b2,
                      float* __restrict__ out) {
  __shared__ float h[256];
  int t = threadIdx.x;
  float s = b1[t];
  for (int k = 0; k < 512; ++k) s += pooled[k] * W1[t * 512 + k];
  h[t] = fmaxf(s, 0.0f);
  __syncthreads();
  if (t < 16) {
    float o = b2[t];
    for (int k = 0; k < 256; ++k) o += h[k] * W2[t * 256 + k];
    out[t] = o;
  }
}

// ------------------------------ host driver --------------------------------

extern "C" void kernel_launch(void* const* d_in, const int* in_sizes, int n_in,
                              void* d_out, int out_size, void* d_ws, size_t ws_size,
                              hipStream_t stream) {
  (void)n_in; (void)out_size; (void)ws_size;
  const float* x_author = (const float*)d_in[0];
  const float* x_paper  = (const float*)d_in[1];
  const int*   w_src = (const int*)d_in[2];
  const int*   w_dst = (const int*)d_in[3];
  const int*   b_src = (const int*)d_in[4];
  const int*   b_dst = (const int*)d_in[5];
  const float* c1w_Wl = (const float*)d_in[6];
  const float* c1w_bl = (const float*)d_in[7];
  const float* c1w_Wr = (const float*)d_in[8];
  const float* c1b_Wl = (const float*)d_in[9];
  const float* c1b_bl = (const float*)d_in[10];
  const float* c1b_Wr = (const float*)d_in[11];
  const float* c2w_Wl = (const float*)d_in[12];
  const float* c2w_bl = (const float*)d_in[13];
  const float* c2w_Wr = (const float*)d_in[14];
  const float* c2b_Wl = (const float*)d_in[15];
  const float* c2b_bl = (const float*)d_in[16];
  const float* c2b_Wr = (const float*)d_in[17];
  const float* skipA_W = (const float*)d_in[18];
  const float* skipA_b = (const float*)d_in[19];
  const float* skipP_W = (const float*)d_in[20];
  const float* skipP_b = (const float*)d_in[21];
  const float* cls_W1 = (const float*)d_in[22];
  const float* cls_b1 = (const float*)d_in[23];
  const float* cls_W2 = (const float*)d_in[24];
  const float* cls_b2 = (const float*)d_in[25];
  float* out = (float*)d_out;

  const long E = (long)in_sizes[2];

  // ---- workspace bump allocator ----
  char* ws = (char*)d_ws;
  size_t off = 0;
  auto alloc = [&](size_t bytes) -> void* {
    off = (off + 255) & ~(size_t)255;
    void* p = ws + off;
    off += bytes;
    return p;
  };

  __bf16* bw_c1w_Wl = (__bf16*)alloc((size_t)H_D * IN_D * 2);
  __bf16* bw_c1w_Wr = (__bf16*)alloc((size_t)H_D * IN_D * 2);
  __bf16* bw_c1b_Wl = (__bf16*)alloc((size_t)H_D * IN_D * 2);
  __bf16* bw_c1b_Wr = (__bf16*)alloc((size_t)H_D * IN_D * 2);
  __bf16* bw_c2w_Wl = (__bf16*)alloc((size_t)H_D * H_D * 2);
  __bf16* bw_c2w_Wr = (__bf16*)alloc((size_t)H_D * H_D * 2);
  __bf16* bw_c2b_Wl = (__bf16*)alloc((size_t)H_D * H_D * 2);
  __bf16* bw_c2b_Wr = (__bf16*)alloc((size_t)H_D * H_D * 2);
  __bf16* bw_skipA  = (__bf16*)alloc((size_t)H_D * IN_D * 2);
  __bf16* bw_skipP  = (__bf16*)alloc((size_t)H_D * IN_D * 2);
  float*  cnt_p = (float*)alloc((size_t)NP_N * 4);
  float*  cnt_a = (float*)alloc((size_t)NA_N * 4);
  float*  agg   = (float*)alloc((size_t)NP_N * H_D * 4);   // reused 4x
  __bf16* h1_p  = (__bf16*)alloc((size_t)NP_N * H_D * 2);
  __bf16* h1_a  = (__bf16*)alloc((size_t)NA_N * H_D * 2);
  __bf16* h2_p  = (__bf16*)alloc((size_t)NP_N * H_D * 2);
  __bf16* h2_a  = (__bf16*)alloc((size_t)NA_N * H_D * 2);
  float*  pooled = (float*)alloc(512 * 4);

  auto gs = [](long n, int b) { return (unsigned)((n + b - 1) / b); };

  // ---- weights fp32 -> bf16 ----
  const long WS = (long)H_D * IN_D, WH = (long)H_D * H_D;
  k_f32_to_bf16<<<gs(WS, 256), 256, 0, stream>>>(c1w_Wl, bw_c1w_Wl, WS);
  k_f32_to_bf16<<<gs(WS, 256), 256, 0, stream>>>(c1w_Wr, bw_c1w_Wr, WS);
  k_f32_to_bf16<<<gs(WS, 256), 256, 0, stream>>>(c1b_Wl, bw_c1b_Wl, WS);
  k_f32_to_bf16<<<gs(WS, 256), 256, 0, stream>>>(c1b_Wr, bw_c1b_Wr, WS);
  k_f32_to_bf16<<<gs(WH, 256), 256, 0, stream>>>(c2w_Wl, bw_c2w_Wl, WH);
  k_f32_to_bf16<<<gs(WH, 256), 256, 0, stream>>>(c2w_Wr, bw_c2w_Wr, WH);
  k_f32_to_bf16<<<gs(WH, 256), 256, 0, stream>>>(c2b_Wl, bw_c2b_Wl, WH);
  k_f32_to_bf16<<<gs(WH, 256), 256, 0, stream>>>(c2b_Wr, bw_c2b_Wr, WH);
  k_f32_to_bf16<<<gs(WS, 256), 256, 0, stream>>>(skipA_W, bw_skipA, WS);
  k_f32_to_bf16<<<gs(WS, 256), 256, 0, stream>>>(skipP_W, bw_skipP, WS);

  // ---- degree counts (same edge lists reused in both layers) ----
  k_zero_f32<<<gs(NP_N, 256), 256, 0, stream>>>(cnt_p, NP_N);
  k_zero_f32<<<gs(NA_N, 256), 256, 0, stream>>>(cnt_a, NA_N);
  k_count<<<gs(E, 256), 256, 0, stream>>>(w_dst, E, cnt_p);
  k_count<<<gs(E, 256), 256, 0, stream>>>(b_dst, E, cnt_a);

  // ---- layer 1, paper dst: h1_p = relu(mean(x_author)@Wl^T + bl + x_paper@Wr^T) ----
  k_zero_f32<<<gs(NP_N * IN_D, 256), 256, 0, stream>>>(agg, NP_N * IN_D);
  k_scatter_add<<<gs(E * (IN_D / 4), 256), 256, 0, stream>>>(x_author, 1, IN_D, w_src, w_dst, E, agg);
  k_div_rows<<<gs(NP_N * IN_D, 256), 256, 0, stream>>>(agg, cnt_p, NP_N, IN_D);
  k_sage_gemm<IN_D, 1, IN_D, 1, 0, 0><<<(unsigned)(NP_N / 16), 32, 0, stream>>>(
      agg, bw_c1w_Wl, x_paper, bw_c1w_Wr, nullptr, nullptr,
      c1w_bl, nullptr, h1_p, 1);

  // ---- layer 1, author dst ----
  k_zero_f32<<<gs(NA_N * IN_D, 256), 256, 0, stream>>>(agg, NA_N * IN_D);
  k_scatter_add<<<gs(E * (IN_D / 4), 256), 256, 0, stream>>>(x_paper, 1, IN_D, b_src, b_dst, E, agg);
  k_div_rows<<<gs(NA_N * IN_D, 256), 256, 0, stream>>>(agg, cnt_a, NA_N, IN_D);
  k_sage_gemm<IN_D, 1, IN_D, 1, 0, 0><<<(unsigned)(NA_N / 16), 32, 0, stream>>>(
      agg, bw_c1b_Wl, x_author, bw_c1b_Wr, nullptr, nullptr,
      c1b_bl, nullptr, h1_a, 1);

  // ---- layer 2, paper dst: relu(mean(h1_a)@Wl^T + bl + h1_p@Wr^T + x_paper@skipP^T + skipP_b) ----
  k_zero_f32<<<gs(NP_N * H_D, 256), 256, 0, stream>>>(agg, NP_N * H_D);
  k_scatter_add<<<gs(E * (H_D / 4), 256), 256, 0, stream>>>(h1_a, 0, H_D, w_src, w_dst, E, agg);
  k_div_rows<<<gs(NP_N * H_D, 256), 256, 0, stream>>>(agg, cnt_p, NP_N, H_D);
  k_sage_gemm<H_D, 1, H_D, 0, IN_D, 1><<<(unsigned)(NP_N / 16), 32, 0, stream>>>(
      agg, bw_c2w_Wl, h1_p, bw_c2w_Wr, x_paper, bw_skipP,
      c2w_bl, skipP_b, h2_p, 1);

  // ---- layer 2, author dst ----
  k_zero_f32<<<gs(NA_N * H_D, 256), 256, 0, stream>>>(agg, NA_N * H_D);
  k_scatter_add<<<gs(E * (H_D / 4), 256), 256, 0, stream>>>(h1_p, 0, H_D, b_src, b_dst, E, agg);
  k_div_rows<<<gs(NA_N * H_D, 256), 256, 0, stream>>>(agg, cnt_a, NA_N, H_D);
  k_sage_gemm<H_D, 1, H_D, 0, IN_D, 1><<<(unsigned)(NA_N / 16), 32, 0, stream>>>(
      agg, bw_c2b_Wl, h1_a, bw_c2b_Wr, x_author, bw_skipA,
      c2b_bl, skipA_b, h2_a, 1);

  // ---- global mean pool per node type + classifier MLP ----
  k_col_mean<<<256, 256, 0, stream>>>(h2_a, NA_N, 1.0f / (float)NA_N, pooled);
  k_col_mean<<<256, 256, 0, stream>>>(h2_p, NP_N, 1.0f / (float)NP_N, pooled + 256);
  k_mlp<<<1, 256, 0, stream>>>(pooled, cls_W1, cls_b1, cls_W2, cls_b2, out);
}
